// SynthesisBlock_62414464745790
// MI455X (gfx1250) — compile-verified, matrix operand order checked by reading, not run on hardware
//
#include <hip/hip_runtime.h>
#include <hip/hip_bf16.h>

// ---------------------------------------------------------------------------
// StyleGAN2 synthesis block on gfx1250 (MI455X).
// Convs run as implicit GEMMs on v_wmma_f32_16x16x32_bf16 (f32 accumulate).
// Zero-padded NHWC bf16 staging keeps the GEMM inner loop branch-free.
// Wave tile = 32 pixels x 32 couts: 8x global_load_b128 + 4x v_wmma per
// K-step (2 loads per wmma, 4 independent accumulators -> no hazard nops).
// ---------------------------------------------------------------------------

typedef __attribute__((ext_vector_type(16))) __bf16 v16bf;
typedef __attribute__((ext_vector_type(8)))  __bf16 v8bf;
typedef __attribute__((ext_vector_type(8)))  float  v8f;

union BF16x16 { v16bf v; v8bf h[2]; };

#define NB   8     // batch
#define CI   512   // in channels
#define CO   512   // out channels
#define HH   32    // input H=W
#define WD   512   // latent dim
#define DW   98    // dilated/padded stage-0 width (>= 97 for 32-wide tile overhang)
#define DH   67    // dilated/padded stage-0 height
#define PW   66    // padded stage-1 grid (64 + 1 halo each side)

__device__ __constant__ const float RC    = 0.0147314f;   // 1/sqrt(9*512)
__device__ __constant__ const float MODRC = 0.0441942f;   // 1/sqrt(512)
#define SQRT2F 1.41421356237f

// --------------------------- style: s = y @ (W*modrc) + b + 1 ---------------
__global__ void style_k(const float* __restrict__ w, const float* __restrict__ mw,
                        const float* __restrict__ mb, float* __restrict__ s) {
  int b  = blockIdx.y;
  int ci = blockIdx.x * 256 + threadIdx.x;
  float acc = 0.f;
  for (int d = 0; d < WD; ++d) acc += w[b * WD + d] * mw[d * CI + ci];
  s[b * CI + ci] = acc * MODRC + mb[ci] + 1.0f;
}

// --------------------------- demod: d = rsqrt(sum((w*rc*s)^2)+eps) ----------
__global__ void demod_k(const float* __restrict__ cw, const float* __restrict__ s,
                        float* __restrict__ dm) {
  __shared__ float red[256];
  int b    = blockIdx.y;
  int co   = blockIdx.x * 64 + (threadIdx.x & 63);
  int part = threadIdx.x >> 6;
  float acc = 0.f;
  for (int ci = part * 128; ci < part * 128 + 128; ++ci) {
    float sv = s[b * CI + ci];
    for (int t = 0; t < 9; ++t) {
      float v = cw[((size_t)t * CI + ci) * CO + co] * sv;
      acc += v * v;
    }
  }
  red[threadIdx.x] = acc;
  __syncthreads();
  if (part == 0) {
    float sum = red[threadIdx.x] + red[threadIdx.x + 64] +
                red[threadIdx.x + 128] + red[threadIdx.x + 192];
    dm[b * CO + co] = rsqrtf(sum * (RC * RC) + 1e-8f);
  }
}

// ------------------- pack demodulated weights -> bf16 [b][tap][co][ci] ------
__global__ void pack_k(const float* __restrict__ cw, const float* __restrict__ s,
                       const float* __restrict__ dm, __bf16* __restrict__ ww) {
  int bt = blockIdx.y, b = bt / 9, tap = bt % 9;
  int idx = blockIdx.x * 256 + threadIdx.x;
  int ci = idx & 511, co = idx >> 9;
  float v = cw[((size_t)tap * CI + ci) * CO + co] * RC * s[b * CI + ci] * dm[b * CO + co];
  ww[((size_t)bt * CO + co) * CI + ci] = (__bf16)v;
}

// -------- x (NCHW f32) -> zero-dilated, zero-padded NHWC bf16 [b][67][98][ci]
// Dilated coords: value at (p,q) is x[(p-2)/2][(q-2)/2] iff p,q even in [2,64].
__global__ void xdil_k(const float* __restrict__ x, __bf16* __restrict__ xd) {
  size_t idx = (size_t)blockIdx.x * 256 + threadIdx.x;  // NB*DH*DW*CI
  int ci = idx & 511;
  int t  = (int)(idx >> 9);
  int q  = t % DW; t /= DW;
  int p  = t % DH;
  int b  = t / DH;
  float v = 0.f;
  if (!(p & 1) && p >= 2 && p <= 64 && !(q & 1) && q >= 2 && q <= 64) {
    int iy = (p - 2) >> 1, ix = (q - 2) >> 1;
    v = x[(((size_t)b * CI + ci) * HH + iy) * HH + ix];
  }
  xd[idx] = (__bf16)v;
}

// ------------- stage0: transposed conv stride2 as branch-free WMMA GEMM -----
// u[b][oy][ox][co] bf16 on the 65x65 grid. Wave tile: 32 pixels x 32 couts.
__global__ void upconv0_k(const __bf16* __restrict__ xd, const __bf16* __restrict__ ww,
                          __bf16* __restrict__ u) {
  int bx  = blockIdx.x;                 // b*65*3 + oy*3 + oxt
  int oxt = bx % 3; bx /= 3;
  int oy  = bx % 65; int b = bx / 65;
  int cog  = blockIdx.y;                // 0..1
  int wave = threadIdx.x >> 5, lane = threadIdx.x & 31;
  int half = lane >> 4, l15 = lane & 15;
  int co0  = cog * 256 + wave * 32;
  int ox0  = oxt * 32;
  v8f acc00 = {}, acc01 = {}, acc10 = {}, acc11 = {};
  const __bf16* wb = ww + (size_t)(b * 9) * CO * CI;
  for (int kh = 0; kh < 3; ++kh) {
    int p = oy + kh;
    if ((p & 1) || p < 2 || p > 64) continue;        // all-zero dilated rows
    const __bf16* rowb = xd + (size_t)(b * DH + p) * DW * CI;
    for (int kw = 0; kw < 3; ++kw) {
      const __bf16* a0base = rowb + (size_t)(ox0 + l15 + kw) * CI + half * 8;
      const __bf16* a1base = a0base + (size_t)16 * CI;
      const __bf16* b0base = wb + (size_t)(kh * 3 + kw) * CO * CI +
                             (size_t)(co0 + l15) * CI + half * 8;
      const __bf16* b1base = b0base + (size_t)16 * CI;
      #pragma unroll 2
      for (int cic = 0; cic < 16; ++cic) {
        int k0 = cic * 32;
        BF16x16 a0, a1, b0, b1;
        a0.h[0] = *(const v8bf*)(a0base + k0);
        a0.h[1] = *(const v8bf*)(a0base + k0 + 16);
        a1.h[0] = *(const v8bf*)(a1base + k0);
        a1.h[1] = *(const v8bf*)(a1base + k0 + 16);
        b0.h[0] = *(const v8bf*)(b0base + k0);
        b0.h[1] = *(const v8bf*)(b0base + k0 + 16);
        b1.h[0] = *(const v8bf*)(b1base + k0);
        b1.h[1] = *(const v8bf*)(b1base + k0 + 16);
        acc00 = __builtin_amdgcn_wmma_f32_16x16x32_bf16(
            false, a0.v, false, b0.v, (short)0, acc00, false, false);
        acc01 = __builtin_amdgcn_wmma_f32_16x16x32_bf16(
            false, a0.v, false, b1.v, (short)0, acc01, false, false);
        acc10 = __builtin_amdgcn_wmma_f32_16x16x32_bf16(
            false, a1.v, false, b0.v, (short)0, acc10, false, false);
        acc11 = __builtin_amdgcn_wmma_f32_16x16x32_bf16(
            false, a1.v, false, b1.v, (short)0, acc11, false, false);
      }
    }
  }
  // D layout: lane -> n = lane&15, m = 8*(lane>>4) + v
  for (int v = 0; v < 8; ++v) {
    int oxA = ox0 + half * 8 + v;        // M rows 0..15  (acc00/acc01)
    int oxB = oxA + 16;                  // M rows 16..31 (acc10/acc11)
    if (oxA < 65) {
      size_t base = ((size_t)(b * 65 + oy) * 65 + oxA) * CO;
      u[base + co0 + l15]      = (__bf16)acc00[v];
      u[base + co0 + 16 + l15] = (__bf16)acc01[v];
    }
    if (oxB < 65) {
      size_t base = ((size_t)(b * 65 + oy) * 65 + oxB) * CO;
      u[base + co0 + l15]      = (__bf16)acc10[v];
      u[base + co0 + 16 + l15] = (__bf16)acc11[v];
    }
  }
}

// -------- blur + noise + bias + lrelu*sqrt2 -> SAME-padded NHWC bf16 --------
// Writes full 66x66 padded grid; border pixels get zeros.
__global__ void blur0_k(const __bf16* __restrict__ u, const float* __restrict__ noise,
                        const float* __restrict__ ns, const float* __restrict__ bias,
                        __bf16* __restrict__ yp) {
  size_t idx = (size_t)blockIdx.x * 256 + threadIdx.x;  // NB*PW*PW*(CI/8)
  int cg = idx & 63;
  int t  = (int)(idx >> 6);
  int px = t % PW; t /= PW;
  int py = t % PW;
  int b  = t / PW;
  int c0 = cg * 8;
  v8bf out = {};
  if (py >= 1 && py <= 64 && px >= 1 && px <= 64) {
    int oy = py - 1, ox = px - 1;
    float acc[8] = {0, 0, 0, 0, 0, 0, 0, 0};
    const float kt[4] = {1.f, 3.f, 3.f, 1.f};
    for (int i = 0; i < 4; ++i) {
      int sy = oy - 1 + i; if (sy < 0 || sy > 64) continue;
      for (int j = 0; j < 4; ++j) {
        int sx = ox - 1 + j; if (sx < 0 || sx > 64) continue;
        float kv = kt[i] * kt[j] * 0.0625f;   // outer([1,3,3,1])/64 * 4
        v8bf uv = *(const v8bf*)(u + ((size_t)(b * 65 + sy) * 65 + sx) * CO + c0);
        #pragma unroll
        for (int c = 0; c < 8; ++c) acc[c] += (float)uv[c] * kv;
      }
    }
    float nz = noise[((size_t)b * 64 + oy) * 64 + ox] * ns[0];
    #pragma unroll
    for (int c = 0; c < 8; ++c) {
      float v = acc[c] + nz + bias[c0 + c];
      v = (v > 0.f ? v : 0.2f * v) * SQRT2F;
      out[c] = (__bf16)v;
    }
  }
  *(v8bf*)(yp + ((size_t)(b * PW + py) * PW + px) * CI + c0) = out;
}

// ------------- stage1: 3x3 SAME conv as branch-free WMMA GEMM + epilogue ----
__global__ void conv1_k(const __bf16* __restrict__ yp, const __bf16* __restrict__ ww,
                        const float* __restrict__ noise, const float* __restrict__ ns,
                        const float* __restrict__ bias, float* __restrict__ out) {
  int bx  = blockIdx.x;                 // b*64*2 + oy*2 + oxt
  int oxt = bx & 1; bx >>= 1;
  int oy  = bx & 63; int b = bx >> 6;
  int cog  = blockIdx.y;                // 0..1
  int wave = threadIdx.x >> 5, lane = threadIdx.x & 31;
  int half = lane >> 4, l15 = lane & 15;
  int co0  = cog * 256 + wave * 32;
  int ox0  = oxt * 32;
  v8f acc00 = {}, acc01 = {}, acc10 = {}, acc11 = {};
  const __bf16* wb = ww + (size_t)(b * 9) * CO * CI;
  for (int kh = 0; kh < 3; ++kh) {
    const __bf16* rowb = yp + (size_t)(b * PW + oy + kh) * PW * CI;   // padded row
    for (int kw = 0; kw < 3; ++kw) {
      const __bf16* a0base = rowb + (size_t)(ox0 + l15 + kw) * CI + half * 8;
      const __bf16* a1base = a0base + (size_t)16 * CI;
      const __bf16* b0base = wb + (size_t)(kh * 3 + kw) * CO * CI +
                             (size_t)(co0 + l15) * CI + half * 8;
      const __bf16* b1base = b0base + (size_t)16 * CI;
      #pragma unroll 2
      for (int cic = 0; cic < 16; ++cic) {
        int k0 = cic * 32;
        BF16x16 a0, a1, b0, b1;
        a0.h[0] = *(const v8bf*)(a0base + k0);
        a0.h[1] = *(const v8bf*)(a0base + k0 + 16);
        a1.h[0] = *(const v8bf*)(a1base + k0);
        a1.h[1] = *(const v8bf*)(a1base + k0 + 16);
        b0.h[0] = *(const v8bf*)(b0base + k0);
        b0.h[1] = *(const v8bf*)(b0base + k0 + 16);
        b1.h[0] = *(const v8bf*)(b1base + k0);
        b1.h[1] = *(const v8bf*)(b1base + k0 + 16);
        acc00 = __builtin_amdgcn_wmma_f32_16x16x32_bf16(
            false, a0.v, false, b0.v, (short)0, acc00, false, false);
        acc01 = __builtin_amdgcn_wmma_f32_16x16x32_bf16(
            false, a0.v, false, b1.v, (short)0, acc01, false, false);
        acc10 = __builtin_amdgcn_wmma_f32_16x16x32_bf16(
            false, a1.v, false, b0.v, (short)0, acc10, false, false);
        acc11 = __builtin_amdgcn_wmma_f32_16x16x32_bf16(
            false, a1.v, false, b1.v, (short)0, acc11, false, false);
      }
    }
  }
  float nsv = ns[0];
  float bv0 = bias[co0 + l15], bv1 = bias[co0 + 16 + l15];
  for (int v = 0; v < 8; ++v) {
    int oxA = ox0 + half * 8 + v;
    int oxB = oxA + 16;
    float nzA = noise[((size_t)b * 64 + oy) * 64 + oxA] * nsv;
    float nzB = noise[((size_t)b * 64 + oy) * 64 + oxB] * nsv;
    float t00 = acc00[v] + nzA + bv0;
    float t01 = acc01[v] + nzA + bv1;
    float t10 = acc10[v] + nzB + bv0;
    float t11 = acc11[v] + nzB + bv1;
    t00 = (t00 > 0.f ? t00 : 0.2f * t00) * SQRT2F;
    t01 = (t01 > 0.f ? t01 : 0.2f * t01) * SQRT2F;
    t10 = (t10 > 0.f ? t10 : 0.2f * t10) * SQRT2F;
    t11 = (t11 > 0.f ? t11 : 0.2f * t11) * SQRT2F;
    out[(((size_t)b * CO + co0 + l15) * 64 + oy) * 64 + oxA]      = t00;
    out[(((size_t)b * CO + co0 + 16 + l15) * 64 + oy) * 64 + oxA] = t01;
    out[(((size_t)b * CO + co0 + l15) * 64 + oy) * 64 + oxB]      = t10;
    out[(((size_t)b * CO + co0 + 16 + l15) * 64 + oy) * 64 + oxB] = t11;
  }
}

// ---------------------------------------------------------------------------
extern "C" void kernel_launch(void* const* d_in, const int* in_sizes, int n_in,
                              void* d_out, int out_size, void* d_ws, size_t ws_size,
                              hipStream_t stream) {
  const float* x       = (const float*)d_in[0];
  const float* w0      = (const float*)d_in[1];
  const float* w1      = (const float*)d_in[2];
  const float* conv0_w = (const float*)d_in[3];
  const float* mod0_w  = (const float*)d_in[4];
  const float* mod0_b  = (const float*)d_in[5];
  const float* conv1_w = (const float*)d_in[6];
  const float* mod1_w  = (const float*)d_in[7];
  const float* mod1_b  = (const float*)d_in[8];
  const float* noise0  = (const float*)d_in[9];
  const float* ns0     = (const float*)d_in[10];
  const float* bias0   = (const float*)d_in[11];
  const float* noise1  = (const float*)d_in[12];
  const float* ns1     = (const float*)d_in[13];
  const float* bias1   = (const float*)d_in[14];
  float* out = (float*)d_out;

  char* ws = (char*)d_ws;
  size_t off = 0;
  auto carve = [&](size_t bytes) { char* p = ws + off; off += (bytes + 255) & ~(size_t)255; return p; };
  float*  s0   = (float*)carve((size_t)NB * CI * 4);
  float*  s1   = (float*)carve((size_t)NB * CI * 4);
  float*  dm0  = (float*)carve((size_t)NB * CO * 4);
  float*  dm1  = (float*)carve((size_t)NB * CO * 4);
  __bf16* ww0  = (__bf16*)carve((size_t)NB * 9 * CO * CI * 2);   // 37.7 MB
  __bf16* ww1  = (__bf16*)carve((size_t)NB * 9 * CO * CI * 2);   // 37.7 MB
  __bf16* xdil = (__bf16*)carve((size_t)NB * DH * DW * CI * 2);  // 53.8 MB
  __bf16* u65  = (__bf16*)carve((size_t)NB * 65 * 65 * CO * 2);  // 34.6 MB
  __bf16* y0p  = (__bf16*)carve((size_t)NB * PW * PW * CI * 2);  // 35.7 MB

  dim3 blk(256);
  // styles + demod + weight packing
  style_k<<<dim3(CI / 256, NB), blk, 0, stream>>>(w0, mod0_w, mod0_b, s0);
  style_k<<<dim3(CI / 256, NB), blk, 0, stream>>>(w1, mod1_w, mod1_b, s1);
  demod_k<<<dim3(CO / 64, NB), blk, 0, stream>>>(conv0_w, s0, dm0);
  demod_k<<<dim3(CO / 64, NB), blk, 0, stream>>>(conv1_w, s1, dm1);
  pack_k<<<dim3((CO * CI) / 256, NB * 9), blk, 0, stream>>>(conv0_w, s0, dm0, ww0);
  pack_k<<<dim3((CO * CI) / 256, NB * 9), blk, 0, stream>>>(conv1_w, s1, dm1, ww1);
  // zero-dilated padded input
  size_t ndil = (size_t)NB * DH * DW * CI;
  xdil_k<<<dim3((unsigned)((ndil + 255) / 256)), blk, 0, stream>>>(x, xdil);
  // stage 0: transposed conv (WMMA) -> 65x65 bf16
  upconv0_k<<<dim3(NB * 65 * 3, 2), blk, 0, stream>>>(xdil, ww0, u65);
  // blur + noise + bias + lrelu -> padded 66x66 bf16
  size_t nblur = (size_t)NB * PW * PW * (CI / 8);
  blur0_k<<<dim3((unsigned)(nblur / 256)), blk, 0, stream>>>(u65, noise0, ns0, bias0, y0p);
  // stage 1: 3x3 conv (WMMA) + epilogue -> NCHW f32
  conv1_k<<<dim3(NB * 64 * 2, 2), blk, 0, stream>>>(y0p, ww1, noise1, ns1, bias1, out);
}
